// BindingSiteGAT_49735721288426
// MI455X (gfx1250) — matrix operand order repaired and checked by reading, not compiled, not run on hardware
//
#include <hip/hip_runtime.h>
#include <hip/hip_bf16.h>

#define NNODES   20000
#define MPAD     20032               // 1252 tiles of 16; divisible by 2- and 4-tile blocks
#define MTILES   (MPAD / 16)
#define NEDGES   160000
#define ETOT     (NEDGES + NNODES)   // with self loops
#define GAT_SLOPE 0.2f
#define ACT_SLOPE 0.15f

typedef __attribute__((ext_vector_type(16))) _Float16 v16h;
typedef __attribute__((ext_vector_type(8)))  _Float16 v8h;
typedef __attribute__((ext_vector_type(8)))  float    v8f;
typedef __attribute__((ext_vector_type(4)))  unsigned u32x4;
typedef __attribute__((ext_vector_type(8)))  int      i32x8;
typedef __attribute__((ext_vector_type(4)))  int      i32x4;

static __device__ __forceinline__ float leaky(float v, float s) {
    return v > 0.0f ? v : s * v;
}

// float atomic max via signed/unsigned int trick (init to -inf)
static __device__ __forceinline__ void atomicMaxFloat(float* addr, float val) {
    if (val >= 0.0f) {
        atomicMax((int*)addr, __float_as_int(val));
    } else {
        atomicMin((unsigned int*)addr, (unsigned int)__float_as_int(val));
    }
}

// generic (shared) pointer -> raw LDS byte offset for TDM descriptors (runtime only)
static __device__ __forceinline__ unsigned ldsOff(const void* p) {
    return (unsigned)(unsigned long long)
        (__attribute__((address_space(3))) const char*)(const char*)p;
}

// ---- Tensor Data Mover: 2-D tile load Global->LDS (D# per CDNA5 ISA 8.3/8.4) ----
// data_size = 2 bytes; tile = tile_x (halfs) x tile_y (rows); row stride = stride halfs.
static __device__ __forceinline__ void tdm_load_2d(unsigned lds_byte_off, const void* gptr,
                                                   unsigned tile_x, unsigned tile_y,
                                                   unsigned tensor_x, unsigned tensor_y,
                                                   unsigned stride) {
    unsigned long long ga = (unsigned long long)(size_t)gptr;
    u32x4 g0;
    g0[0] = 1u;                                   // count=1, user descriptor
    g0[1] = lds_byte_off;                         // lds_addr
    g0[2] = (unsigned)ga;                         // global_addr[31:0]
    g0[3] = (unsigned)(ga >> 32) | (2u << 30);    // global_addr[56:32] | type=2
    i32x8 g1;
    g1[0] = (int)(1u << 16);                      // wg_mask=0, data_size=1 (2 bytes)
    g1[1] = (int)((tensor_x & 0xFFFFu) << 16);    // tensor_dim0[15:0] @bit48
    g1[2] = (int)(((tensor_x >> 16) & 0xFFFFu) | ((tensor_y & 0xFFFFu) << 16));
    g1[3] = (int)(((tensor_y >> 16) & 0xFFFFu) | ((tile_x & 0xFFFFu) << 16));
    g1[4] = (int)(tile_y & 0xFFFFu);              // tile_dim1; tile_dim2=0
    g1[5] = (int)stride;                          // tensor_dim0_stride[31:0]
    g1[6] = (int)((stride & 0xFFFFu) << 16);      // dim0_stride[47:32]=0 | dim1_stride[15:0]
    g1[7] = (int)(stride >> 16);                  // dim1_stride[47:16]
    i32x4 z4 = {0, 0, 0, 0};
#if __clang_major__ >= 23
    i32x8 z8 = {0, 0, 0, 0, 0, 0, 0, 0};
    __builtin_amdgcn_tensor_load_to_lds(g0, g1, z4, z4, z8, 0);
#else
    __builtin_amdgcn_tensor_load_to_lds(g0, g1, z4, z4, 0);
#endif
}

// ---------------- edge list with self loops ----------------
__global__ void k_build_edges(const int* __restrict__ ei, int* __restrict__ src,
                              int* __restrict__ dst) {
    int t = blockIdx.x * blockDim.x + threadIdx.x;
    if (t >= ETOT) return;
    if (t < NEDGES) { src[t] = ei[t]; dst[t] = ei[NEDGES + t]; }
    else            { int n = t - NEDGES; src[t] = n; dst[t] = n; }
}

// ---------------- f32 -> f16 row pack, zero pad rows to MPAD and K to Kp ----------------
__global__ void k_pack_f16(const float* __restrict__ A, _Float16* __restrict__ Ah,
                           int srcRows, int K, int Kp) {
    long long t = (long long)blockIdx.x * blockDim.x + threadIdx.x;
    long long total = (long long)MPAD * Kp;
    if (t >= total) return;
    int k = (int)(t % Kp);
    long long r = t / Kp;
    Ah[t] = (r < srcRows && k < K) ? (_Float16)A[r * K + k] : (_Float16)0.0f;
}

// W [K x Nc] f32 -> Wt [Nc x Kp] f16 (transposed, K zero-padded)
__global__ void k_transpose_f16(const float* __restrict__ W, _Float16* __restrict__ Wt,
                                int K, int Kp, int Nc) {
    long long t = (long long)blockIdx.x * blockDim.x + threadIdx.x;
    long long total = (long long)Nc * Kp;
    if (t >= total) return;
    int k = (int)(t % Kp);
    int n = (int)(t / Kp);
    Wt[t] = (k < K) ? (_Float16)W[(long long)k * Nc + n] : (_Float16)0.0f;
}

// ---------------- WMMA GEMM with TDM-staged, double-buffered LDS tiles ----------------
// C[MPAD x Nc] = A[MPAD x Kp] * Bt[Nc x Kp]^T
// Block = 8 waves = WM m-tiles x WN n-strips (strip = 64 cols); wave tile = 16x64.
// Per K-step (32 halfs): TDM loads A tile (WM*16 rows) and B tile (WN*64 rows) into LDS;
// fragments come from LDS (ds_load_b128), contiguous k = 16*g + s in both A and B.
template<int WM, int WN>
__global__ void k_gemm_wmma(const _Float16* __restrict__ A,
                            const _Float16* __restrict__ Bt,
                            float* __restrict__ C,
                            int Kp, int Nc) {
    constexpr int AROWS = WM * 16;
    constexpr int BROWS = WN * 64;
    // buffer offsets in halfs (no pointer tables -> no LDS addrspacecast in initializers)
    constexpr int offA0 = 0;
    constexpr int offA1 = AROWS * 32;
    constexpr int offB0 = 2 * AROWS * 32;
    constexpr int offB1 = 2 * AROWS * 32 + BROWS * 32;
    __shared__ __align__(64) _Float16 sm[2 * AROWS * 32 + 2 * BROWS * 32];

    const int nbCount = Nc / (64 * WN);
    const int mb = blockIdx.x / nbCount;
    const int nb = blockIdx.x % nbCount;

    const int wid = threadIdx.x >> 5;
    const int wm = wid % WM;
    const int wn = wid / WM;
    const int lane = threadIdx.x & 31;
    const int g  = lane >> 4;
    const int lm = lane & 15;

    const _Float16* Abase = A  + (long long)(mb * AROWS) * Kp;
    const _Float16* Bbase = Bt + (long long)(nb * BROWS) * Kp;
    const unsigned ldsBase = ldsOff(sm);

    v8f acc[4] = {};
    const int nsteps = Kp >> 5;

    // prologue: stage buffer 0
    if (threadIdx.x < 32) {
        tdm_load_2d(ldsBase + offA0 * 2, Abase, 32, AROWS, (unsigned)Kp, MPAD, (unsigned)Kp);
        tdm_load_2d(ldsBase + offB0 * 2, Bbase, 32, BROWS, (unsigned)Kp, (unsigned)Nc, (unsigned)Kp);
    }

    for (int s = 0; s < nsteps; ++s) {
        const int cur = s & 1;
        const bool more = (s + 1) < nsteps;
        if (more && threadIdx.x < 32) {
            const int k0n = (s + 1) * 32;
            const int an = cur ? offA0 : offA1;
            const int bn = cur ? offB0 : offB1;
            tdm_load_2d(ldsBase + an * 2, Abase + k0n, 32, AROWS, (unsigned)Kp, MPAD, (unsigned)Kp);
            tdm_load_2d(ldsBase + bn * 2, Bbase + k0n, 32, BROWS, (unsigned)Kp, (unsigned)Nc, (unsigned)Kp);
        }
        if (threadIdx.x < 32) {
            if (more) __builtin_amdgcn_s_wait_tensorcnt(2);  // current pair done, next in flight
            else      __builtin_amdgcn_s_wait_tensorcnt(0);
        }
        __syncthreads();

        const _Float16* smA = sm + (cur ? offA1 : offA0);
        const _Float16* smB = sm + (cur ? offB1 : offB0);

        // A fragment: lane row = wm*16+lm; halfs [g*16..g*16+15] contiguous
        v16h a;
        {
            const _Float16* ar = smA + (wm * 16 + lm) * 32;
            const v8h a0 = *(const v8h*)(ar + g * 16);
            const v8h a1 = *(const v8h*)(ar + g * 16 + 8);
            #pragma unroll
            for (int i = 0; i < 8; ++i) { a[i] = a0[i]; a[i + 8] = a1[i]; }
        }
        #pragma unroll
        for (int j = 0; j < 4; ++j) {
            // B fragment: lane col = wn*64+j*16+lm; halfs [g*16..g*16+15] contiguous
            const v16h b = *(const v16h*)(smB + (wn * 64 + j * 16 + lm) * 32 + g * 16);
            acc[j] = __builtin_amdgcn_wmma_f32_16x16x32_f16(
                false, a, false, b, (short)0, acc[j], false, false);
        }
        __syncthreads();   // all waves done reading before next-buffer overwrite
    }

    // D layout: VGPR v -> row = v + g*8, col = lm
    const int rowBase = mb * AROWS + wm * 16;
    const int colBase = nb * BROWS + wn * 64;
    #pragma unroll
    for (int j = 0; j < 4; ++j) {
        const int col = colBase + j * 16 + lm;
        #pragma unroll
        for (int v = 0; v < 8; ++v) {
            C[(long long)(rowBase + v + g * 8) * Nc + col] = acc[j][v];
        }
    }
}

// ---------------- attention scores: a_s/a_d [N, H] ----------------
__global__ void k_att_scores(const float* __restrict__ Hf,
                             const float* __restrict__ att_s,
                             const float* __restrict__ att_d,
                             float* __restrict__ as_, float* __restrict__ ad_,
                             int heads, int C) {
    int t = blockIdx.x * blockDim.x + threadIdx.x;
    if (t >= NNODES * heads) return;
    int n = t / heads, h = t % heads;
    const float* hp = Hf + (long long)n * heads * C + (long long)h * C;
    float s = 0.0f, d = 0.0f;
    for (int c = 0; c < C; ++c) { float v = hp[c]; s += v * att_s[h * C + c]; d += v * att_d[h * C + c]; }
    as_[t] = s; ad_[t] = d;
}

// zero agg [N*HC], init m=-inf / den=0 for [N*heads]
__global__ void k_init_layer(float* __restrict__ m, float* __restrict__ den,
                             float* __restrict__ agg, int nH, long long nAgg) {
    long long t = (long long)blockIdx.x * blockDim.x + threadIdx.x;
    if (t < nAgg) agg[t] = 0.0f;
    if (t < nH) { m[t] = -__int_as_float(0x7f800000); den[t] = 0.0f; }
}

__global__ void k_edge_max(const int* __restrict__ src, const int* __restrict__ dst,
                           const float* __restrict__ as_, const float* __restrict__ ad_,
                           float* __restrict__ ev, float* __restrict__ m, int heads) {
    int t = blockIdx.x * blockDim.x + threadIdx.x;
    if (t >= ETOT * heads) return;
    int e = t / heads, h = t % heads;
    float val = as_[src[e] * heads + h] + ad_[dst[e] * heads + h];
    val = leaky(val, GAT_SLOPE);
    ev[t] = val;
    atomicMaxFloat(m + dst[e] * heads + h, val);
}

__global__ void k_edge_exp(const int* __restrict__ dst, float* __restrict__ ev,
                           const float* __restrict__ m, float* __restrict__ den,
                           int heads) {
    int t = blockIdx.x * blockDim.x + threadIdx.x;
    if (t >= ETOT * heads) return;
    int e = t / heads, h = t % heads;
    float w = __expf(ev[t] - m[dst[e] * heads + h]);
    ev[t] = w;
    atomicAdd(den + dst[e] * heads + h, w);
}

__global__ void k_edge_scatter(const int* __restrict__ src, const int* __restrict__ dst,
                               const float* __restrict__ ev, const float* __restrict__ den,
                               const float* __restrict__ Hf, float* __restrict__ agg,
                               int heads, int C) {
    int t = blockIdx.x * blockDim.x + threadIdx.x;
    if (t >= ETOT * heads) return;
    int e = t / heads, h = t % heads;
    int s = src[e], d = dst[e];
    float alpha = ev[t] / den[d * heads + h];
    const float* hs = Hf + (long long)s * heads * C + (long long)h * C;
    float* od = agg + (long long)d * heads * C + (long long)h * C;
    for (int c = 0; c < C; ++c) atomicAdd(od + c, hs[c] * alpha);
}

// bias + LeakyReLU(0.15) -> f16; rows beyond NNODES zero-filled (GEMM M padding)
__global__ void k_finalize_f16(const float* __restrict__ agg, const float* __restrict__ bias,
                               _Float16* __restrict__ out, int HC) {
    long long t = (long long)blockIdx.x * blockDim.x + threadIdx.x;
    if (t >= (long long)MPAD * HC) return;
    long long r = t / HC;
    out[t] = (r < NNODES) ? (_Float16)leaky(agg[t] + bias[t % HC], ACT_SLOPE) : (_Float16)0.0f;
}

__global__ void k_finalize_f32(const float* __restrict__ agg, const float* __restrict__ bias,
                               float* __restrict__ out, int HC) {
    long long t = (long long)blockIdx.x * blockDim.x + threadIdx.x;
    if (t >= (long long)NNODES * HC) return;
    out[t] = leaky(agg[t] + bias[t % HC], ACT_SLOPE);
}

// pre_fc(128->16) -> fc1(16->32)+leaky -> fc2(32->2)
__global__ void k_mlp(const float* __restrict__ H3f,
                      const float* __restrict__ pw, const float* __restrict__ pb,
                      const float* __restrict__ f1w, const float* __restrict__ f1b,
                      const float* __restrict__ f2w, const float* __restrict__ f2b,
                      float* __restrict__ out) {
    int n = blockIdx.x * blockDim.x + threadIdx.x;
    if (n >= NNODES) return;
    const float* h = H3f + (long long)n * 128;
    float t16[16];
    for (int j = 0; j < 16; ++j) t16[j] = pb[j];
    for (int i = 0; i < 128; ++i) {
        float hv = h[i];
        for (int j = 0; j < 16; ++j) t16[j] += hv * pw[i * 16 + j];
    }
    float t32[32];
    for (int j = 0; j < 32; ++j) t32[j] = f1b[j];
    for (int i = 0; i < 16; ++i) {
        float hv = t16[i];
        for (int j = 0; j < 32; ++j) t32[j] += hv * f1w[i * 32 + j];
    }
    for (int j = 0; j < 32; ++j) t32[j] = leaky(t32[j], ACT_SLOPE);
    float o0 = f2b[0], o1 = f2b[1];
    for (int i = 0; i < 32; ++i) { o0 += t32[i] * f2w[i * 2]; o1 += t32[i] * f2w[i * 2 + 1]; }
    out[n * 2 + 0] = o0;
    out[n * 2 + 1] = o1;
}

static inline long long cdiv(long long a, long long b) { return (a + b - 1) / b; }

extern "C" void kernel_launch(void* const* d_in, const int* in_sizes, int n_in,
                              void* d_out, int out_size, void* d_ws, size_t ws_size,
                              hipStream_t stream) {
    (void)in_sizes; (void)n_in; (void)out_size; (void)ws_size;
    const float* x   = (const float*)d_in[0];
    const int*   ei  = (const int*)d_in[1];
    const float* W1  = (const float*)d_in[2];
    const float* as1 = (const float*)d_in[3];
    const float* ad1 = (const float*)d_in[4];
    const float* b1  = (const float*)d_in[5];
    const float* W2  = (const float*)d_in[6];
    const float* as2 = (const float*)d_in[7];
    const float* ad2 = (const float*)d_in[8];
    const float* b2  = (const float*)d_in[9];
    const float* W3  = (const float*)d_in[10];
    const float* as3 = (const float*)d_in[11];
    const float* ad3 = (const float*)d_in[12];
    const float* b3  = (const float*)d_in[13];
    const float* pw  = (const float*)d_in[14];
    const float* pb  = (const float*)d_in[15];
    const float* f1w = (const float*)d_in[16];
    const float* f1b = (const float*)d_in[17];
    const float* f2w = (const float*)d_in[18];
    const float* f2b = (const float*)d_in[19];

    const int Kp1 = 1312;                // 1287 padded to multiple of 32
    char* ws = (char*)d_ws;
    size_t off = 0;
    auto alloc = [&](size_t bytes) { size_t o = off; off = (off + bytes + 255) & ~(size_t)255; return o; };

    size_t o_Xh  = alloc((size_t)MPAD * Kp1 * 2);
    size_t o_W1t = alloc((size_t)512 * Kp1 * 2);
    size_t o_H1  = alloc((size_t)MPAD * 512 * 4);
    size_t o_H1h = alloc((size_t)MPAD * 512 * 2);
    size_t o_W2t = alloc((size_t)1024 * 512 * 2);
    size_t o_H2  = alloc((size_t)MPAD * 1024 * 4);
    size_t o_H2h = alloc((size_t)MPAD * 1024 * 2);
    size_t o_W3t = alloc((size_t)128 * 1024 * 2);
    size_t o_H3  = alloc((size_t)MPAD * 128 * 4);
    size_t o_H3f = alloc((size_t)NNODES * 128 * 4);
    size_t o_as  = alloc((size_t)NNODES * 4 * 4);
    size_t o_ad  = alloc((size_t)NNODES * 4 * 4);
    size_t o_m   = alloc((size_t)NNODES * 4 * 4);
    size_t o_den = alloc((size_t)NNODES * 4 * 4);
    size_t o_ev  = alloc((size_t)ETOT * 4 * 4);
    size_t o_src = alloc((size_t)ETOT * 4);
    size_t o_dst = alloc((size_t)ETOT * 4);
    // AGG aliases [Xh..H2): Xh/W1t dead after gemm1; H1/H1h/W2t dead by layer-2 aggregation.
    size_t o_AGG = 0;

    int* srcp = (int*)(ws + o_src);
    int* dstp = (int*)(ws + o_dst);
    float* AGG = (float*)(ws + o_AGG);
    float* as_ = (float*)(ws + o_as);
    float* ad_ = (float*)(ws + o_ad);
    float* m_  = (float*)(ws + o_m);
    float* den_= (float*)(ws + o_den);
    float* ev  = (float*)(ws + o_ev);

    const int TB = 256;
    k_build_edges<<<cdiv(ETOT, TB), TB, 0, stream>>>(ei, srcp, dstp);

    // ---- layer 1: 1287 -> 4x128, concat ----
    k_pack_f16<<<cdiv((long long)MPAD * Kp1, TB), TB, 0, stream>>>(x, (_Float16*)(ws + o_Xh), NNODES, 1287, Kp1);
    k_transpose_f16<<<cdiv((long long)512 * Kp1, TB), TB, 0, stream>>>(W1, (_Float16*)(ws + o_W1t), 1287, Kp1, 512);
    k_gemm_wmma<2, 4><<<(MTILES / 2) * (512 / 256), 256, 0, stream>>>(
        (const _Float16*)(ws + o_Xh), (const _Float16*)(ws + o_W1t), (float*)(ws + o_H1), Kp1, 512);
    k_att_scores<<<cdiv(NNODES * 4, TB), TB, 0, stream>>>((const float*)(ws + o_H1), as1, ad1, as_, ad_, 4, 128);
    k_init_layer<<<cdiv((long long)NNODES * 512, TB), TB, 0, stream>>>(m_, den_, AGG, NNODES * 4, (long long)NNODES * 512);
    k_edge_max<<<cdiv(ETOT * 4, TB), TB, 0, stream>>>(srcp, dstp, as_, ad_, ev, m_, 4);
    k_edge_exp<<<cdiv(ETOT * 4, TB), TB, 0, stream>>>(dstp, ev, m_, den_, 4);
    k_edge_scatter<<<cdiv(ETOT * 4, TB), TB, 0, stream>>>(srcp, dstp, ev, den_, (const float*)(ws + o_H1), AGG, 4, 128);
    k_finalize_f16<<<cdiv((long long)MPAD * 512, TB), TB, 0, stream>>>(AGG, b1, (_Float16*)(ws + o_H1h), 512);

    // ---- layer 2: 512 -> 4x256, concat ----
    k_transpose_f16<<<cdiv((long long)1024 * 512, TB), TB, 0, stream>>>(W2, (_Float16*)(ws + o_W2t), 512, 512, 1024);
    k_gemm_wmma<2, 4><<<(MTILES / 2) * (1024 / 256), 256, 0, stream>>>(
        (const _Float16*)(ws + o_H1h), (const _Float16*)(ws + o_W2t), (float*)(ws + o_H2), 512, 1024);
    k_att_scores<<<cdiv(NNODES * 4, TB), TB, 0, stream>>>((const float*)(ws + o_H2), as2, ad2, as_, ad_, 4, 256);
    k_init_layer<<<cdiv((long long)NNODES * 1024, TB), TB, 0, stream>>>(m_, den_, AGG, NNODES * 4, (long long)NNODES * 1024);
    k_edge_max<<<cdiv(ETOT * 4, TB), TB, 0, stream>>>(srcp, dstp, as_, ad_, ev, m_, 4);
    k_edge_exp<<<cdiv(ETOT * 4, TB), TB, 0, stream>>>(dstp, ev, m_, den_, 4);
    k_edge_scatter<<<cdiv(ETOT * 4, TB), TB, 0, stream>>>(srcp, dstp, ev, den_, (const float*)(ws + o_H2), AGG, 4, 256);
    k_finalize_f16<<<cdiv((long long)MPAD * 1024, TB), TB, 0, stream>>>(AGG, b2, (_Float16*)(ws + o_H2h), 1024);

    // ---- layer 3: 1024 -> 1x128, mean(1 head) ----
    k_transpose_f16<<<cdiv((long long)128 * 1024, TB), TB, 0, stream>>>(W3, (_Float16*)(ws + o_W3t), 1024, 1024, 128);
    k_gemm_wmma<4, 2><<<(MTILES / 4) * (128 / 128), 256, 0, stream>>>(
        (const _Float16*)(ws + o_H2h), (const _Float16*)(ws + o_W3t), (float*)(ws + o_H3), 1024, 128);
    k_att_scores<<<cdiv(NNODES, TB), TB, 0, stream>>>((const float*)(ws + o_H3), as3, ad3, as_, ad_, 1, 128);
    k_init_layer<<<cdiv((long long)NNODES * 128, TB), TB, 0, stream>>>(m_, den_, AGG, NNODES, (long long)NNODES * 128);
    k_edge_max<<<cdiv(ETOT, TB), TB, 0, stream>>>(srcp, dstp, as_, ad_, ev, m_, 1);
    k_edge_exp<<<cdiv(ETOT, TB), TB, 0, stream>>>(dstp, ev, m_, den_, 1);
    k_edge_scatter<<<cdiv(ETOT, TB), TB, 0, stream>>>(srcp, dstp, ev, den_, (const float*)(ws + o_H3), AGG, 1, 128);
    k_finalize_f32<<<cdiv((long long)NNODES * 128, TB), TB, 0, stream>>>(AGG, b3, (float*)(ws + o_H3f), 128);

    // ---- head MLP ----
    k_mlp<<<cdiv(NNODES, TB), TB, 0, stream>>>((const float*)(ws + o_H3f), pw, pb, f1w, f1b, f2w, f2b, (float*)d_out);
}
